// MultiGPUMoELayer_77567109366040
// MI455X (gfx1250) — matrix operand order repaired and checked by reading, not compiled, not run on hardware
//
#include <hip/hip_runtime.h>
#include <hip/hip_bf16.h>
#include <math.h>

// Problem constants (B=4, S=2048, H=2048, F=8192, E=8, TOP_K=2)
#define TDIM 8192        // B*S tokens
#define HDIM 2048
#define FDIM 8192
#define EXP  8
#define MTILES 64        // max 128-row M tiles per expert = TDIM/128

typedef __attribute__((ext_vector_type(16))) __bf16 v16bf;
typedef __attribute__((ext_vector_type(8)))  float  v8f;

struct U32x8 { uint4 lo, hi; };

static __device__ __forceinline__ v16bf make_frag(uint4 lo, uint4 hi) {
  U32x8 t{lo, hi};
  return __builtin_bit_cast(v16bf, t);
}

// f32 -> bf16 round-to-nearest-even
static __device__ __forceinline__ unsigned short f2bf(float f) {
  unsigned int u = __float_as_uint(f);
  u += 0x7fffu + ((u >> 16) & 1u);
  return (unsigned short)(u >> 16);
}

// Async copy 16B global -> LDS (tracked by ASYNCcnt, bypasses VGPRs).
static __device__ __forceinline__ void async_copy_b128(unsigned lds_off,
                                                       const void* gaddr) {
  asm volatile("global_load_async_to_lds_b128 %0, %1, off"
               :: "v"(lds_off), "v"(gaddr) : "memory");
}
static __device__ __forceinline__ void wait_asynccnt0() {
  asm volatile("s_wait_asynccnt 0x0" ::: "memory");
}

// ---------------------------------------------------------------------------
// 1) Gate: router logits -> softmax -> top-2 -> renorm. Also emit bf16 tokens.
// ---------------------------------------------------------------------------
__global__ __launch_bounds__(256)
void gate_kernel(const float* __restrict__ hidden,
                 const float* __restrict__ gate_w,   // [H, E] row-major
                 float* __restrict__ topw,           // [T, 2]
                 int* __restrict__ topidx,           // [T, 2]
                 unsigned short* __restrict__ tokensBf) // [T, H] bf16
{
  const int t = blockIdx.x;
  const float* row = hidden + (size_t)t * HDIM;
  float acc[EXP];
  #pragma unroll
  for (int e = 0; e < EXP; ++e) acc[e] = 0.f;

  for (int h = threadIdx.x; h < HDIM; h += 256) {
    float x = row[h];
    tokensBf[(size_t)t * HDIM + h] = f2bf(x);
    const float* g = gate_w + (size_t)h * EXP;
    #pragma unroll
    for (int e = 0; e < EXP; ++e) acc[e] += x * g[e];
  }

  __shared__ float red[EXP][256];
  #pragma unroll
  for (int e = 0; e < EXP; ++e) red[e][threadIdx.x] = acc[e];
  __syncthreads();
  for (int s = 128; s > 0; s >>= 1) {
    if (threadIdx.x < (unsigned)s) {
      #pragma unroll
      for (int e = 0; e < EXP; ++e) red[e][threadIdx.x] += red[e][threadIdx.x + s];
    }
    __syncthreads();
  }

  if (threadIdx.x == 0) {
    float p[EXP];
    float mx = red[0][0];
    #pragma unroll
    for (int e = 1; e < EXP; ++e) mx = fmaxf(mx, red[e][0]);
    float sum = 0.f;
    #pragma unroll
    for (int e = 0; e < EXP; ++e) { p[e] = __expf(red[e][0] - mx); sum += p[e]; }
    float inv = 1.f / sum;
    #pragma unroll
    for (int e = 0; e < EXP; ++e) p[e] *= inv;
    // top-2 (ties -> lowest index, matching jax.lax.top_k)
    int i0 = 0;
    #pragma unroll
    for (int e = 1; e < EXP; ++e) if (p[e] > p[i0]) i0 = e;
    int i1 = (i0 == 0) ? 1 : 0;
    #pragma unroll
    for (int e = 0; e < EXP; ++e) if (e != i0 && p[e] > p[i1]) i1 = e;
    float w0 = p[i0], w1 = p[i1];
    float winv = 1.f / (w0 + w1);
    topidx[t * 2 + 0] = i0;  topw[t * 2 + 0] = w0 * winv;
    topidx[t * 2 + 1] = i1;  topw[t * 2 + 1] = w1 * winv;
  }
}

// ---------------------------------------------------------------------------
// 2) Routing: atomics only pick *positions*; every output slot's value is
//    independent of ordering -> deterministic results.
// ---------------------------------------------------------------------------
__global__ void route_init(int* __restrict__ counts, int* __restrict__ cursor) {
  int i = threadIdx.x;
  if (i < EXP) { counts[i] = 0; cursor[i] = 0; }
}

__global__ void route_count(const int* __restrict__ topidx, int* __restrict__ counts) {
  int i = blockIdx.x * blockDim.x + threadIdx.x;
  if (i < 2 * TDIM) atomicAdd(&counts[topidx[i]], 1);
}

__global__ void route_scan(const int* __restrict__ counts, int* __restrict__ baseOff) {
  if (threadIdx.x == 0 && blockIdx.x == 0) {
    int s = 0;
    for (int e = 0; e < EXP; ++e) { baseOff[e] = s; s += counts[e]; }
  }
}

__global__ void route_scatter(const int* __restrict__ topidx,
                              const float* __restrict__ topw,
                              const int* __restrict__ baseOff,
                              int* __restrict__ cursor,
                              int* __restrict__ rowlist,   // [2T] token id per slot
                              float* __restrict__ poswArr, // [2T] gate weight per slot
                              int* __restrict__ tok2pos)   // [T,2] slot of (token,k)
{
  int i = blockIdx.x * blockDim.x + threadIdx.x;
  if (i >= 2 * TDIM) return;
  int t = i >> 1;
  int e = topidx[i];
  int pos = atomicAdd(&cursor[e], 1);
  int p = baseOff[e] + pos;
  rowlist[p] = t;
  poswArr[p] = topw[i];
  tok2pos[i] = p;
}

// ---------------------------------------------------------------------------
// 3) Grouped GEMM (GEMM1 and GEMM2). 128x128 block tile, BK=32, 8 waves,
//    wave tile 64x32, v_wmma_f32_16x16x32_bf16 with f32 accumulate.
//    LDS double-buffered; A tile via GLOBAL_LOAD_ASYNC_TO_LDS_B128 (ASYNCcnt);
//    B tile f32->bf16 converted in registers; one barrier per K-step.
// ---------------------------------------------------------------------------
__global__ __launch_bounds__(256, 2)
void moe_gemm_kernel(const unsigned short* __restrict__ Asrc, // bf16 [rows, K]
                     const float* __restrict__ W,             // [E, K, N] f32
                     const float* __restrict__ bias,          // [E, N]
                     const int* __restrict__ counts,
                     const int* __restrict__ baseOff,
                     const int* __restrict__ rowlist,         // gemm1 gather
                     const float* __restrict__ posw,          // gemm2 scale
                     unsigned short* __restrict__ dstBf,      // gemm1: hmid [2T, N]
                     float* __restrict__ dstF,                // gemm2: partial [2T, N]
                     int K, int N, int isGemm1)
{
  const int e     = blockIdx.y / MTILES;
  const int mtile = blockIdx.y % MTILES;
  const int cnt   = counts[e];
  if (mtile * 128 >= cnt) return;
  const int base  = baseOff[e];
  const int nBlk  = blockIdx.x * 128;

  const int tid  = threadIdx.x;
  const int lane = tid & 31;
  const int wv   = tid >> 5;
  const int wM   = wv & 1;       // 2 M-blocks of 64
  const int wN   = wv >> 1;      // 4 N-blocks of 32
  const int hs   = lane >> 4;    // lane half-select
  const int l16  = lane & 15;

  const float* We = W + (size_t)e * (size_t)K * (size_t)N;

  // Per-thread A staging source row (row = tid/2, 32B half per thread)
  const int rowA  = tid >> 1;
  const int halfA = tid & 1;
  const int gRowA = mtile * 128 + rowA;
  const unsigned short* aRow;
  if (isGemm1) {
    int tok = rowlist[base + (gRowA < cnt ? gRowA : 0)];
    aRow = Asrc + (size_t)tok * (size_t)K;
  } else {
    int p = base + gRowA;
    if (p > 2 * TDIM - 1) p = 2 * TDIM - 1;
    aRow = Asrc + (size_t)p * (size_t)K;
  }

  __shared__ uint4 ldsA[2][512];   // per buf: 128 rows x 32 bf16 (64B/row)
  __shared__ uint4 ldsB[2][512];   // per buf: 128 cols x 32 bf16 (K-major/col)

  const v8f vzero = (v8f){0.f, 0.f, 0.f, 0.f, 0.f, 0.f, 0.f, 0.f};
  v8f acc[4][2];
  #pragma unroll
  for (int ai = 0; ai < 4; ++ai) { acc[ai][0] = vzero; acc[ai][1] = vzero; }

  float4 bReg[4];
  const int nt = K >> 5;   // number of 32-wide K tiles

  // --- helpers (inlined) ---
  auto issueA = [&](int tile, int buf) {
    const uint4* src = (const uint4*)(aRow + tile * 32) + halfA * 2;
    unsigned d0 = (unsigned)(unsigned long long)&ldsA[buf][rowA * 4 + halfA * 2];
    async_copy_b128(d0, src);
    async_copy_b128(d0 + 16, src + 1);
  };
  auto loadB = [&](int tile) {
    int k0 = tile * 32;
    #pragma unroll
    for (int i = 0; i < 4; ++i) {
      int lin = tid + i * 256;           // 1024 float4 = 32 Krows x 32 Ngroups
      int kk = lin >> 5;
      int nf = lin & 31;
      bReg[i] = *(const float4*)(We + (size_t)(k0 + kk) * N + nBlk + nf * 4);
      if (k0 + 64 < K)   // L2 prefetch two tiles ahead (global_prefetch_b8)
        __builtin_prefetch(We + (size_t)(k0 + 64 + kk) * N + nBlk + nf * 4, 0, 1);
    }
  };
  auto storeB = [&](int buf) {
    unsigned short* dst = reinterpret_cast<unsigned short*>(ldsB[buf]);
    #pragma unroll
    for (int i = 0; i < 4; ++i) {
      int lin = tid + i * 256;
      int kk = lin >> 5;
      int nb = (lin & 31) * 4;
      dst[(nb + 0) * 32 + kk] = f2bf(bReg[i].x);
      dst[(nb + 1) * 32 + kk] = f2bf(bReg[i].y);
      dst[(nb + 2) * 32 + kk] = f2bf(bReg[i].z);
      dst[(nb + 3) * 32 + kk] = f2bf(bReg[i].w);
    }
  };
  auto computeTile = [&](int buf) {
    // B fragments: lanes 0-15 K=0..15, lanes 16-31 K=16..31 (contiguous 32B)
    int n0 = wN * 32 + l16;
    v16bf bf0 = make_frag(ldsB[buf][n0 * 4 + hs * 2], ldsB[buf][n0 * 4 + hs * 2 + 1]);
    int n1 = n0 + 16;
    v16bf bf1 = make_frag(ldsB[buf][n1 * 4 + hs * 2], ldsB[buf][n1 * 4 + hs * 2 + 1]);
    // A fragments: lanes 0-15 K=0..15, lanes 16-31 K=16..31 (contiguous 32B)
    #pragma unroll
    for (int ai = 0; ai < 4; ++ai) {
      int m = wM * 64 + ai * 16 + l16;
      v16bf af = make_frag(ldsA[buf][m * 4 + hs * 2], ldsA[buf][m * 4 + hs * 2 + 1]);
      acc[ai][0] = __builtin_amdgcn_wmma_f32_16x16x32_bf16(
          false, af, false, bf0, (short)0, acc[ai][0], false, false);
      acc[ai][1] = __builtin_amdgcn_wmma_f32_16x16x32_bf16(
          false, af, false, bf1, (short)0, acc[ai][1], false, false);
    }
  };

  // --- prologue: fill buffer 0 ---
  issueA(0, 0);
  loadB(0);
  storeB(0);
  wait_asynccnt0();
  __syncthreads();

  // --- main loop: one barrier per K-step ---
  for (int t = 0; t < nt; ++t) {
    const int cur = t & 1;
    const int nxt = cur ^ 1;
    if (t + 1 < nt) {
      issueA(t + 1, nxt);   // async engine fills ldsA[nxt] during compute
      loadB(t + 1);         // global f32 loads in flight during compute
    }
    computeTile(cur);
    if (t + 1 < nt) {
      storeB(nxt);          // cvt + ds_store into the non-read buffer
      wait_asynccnt0();     // own async copies done before barrier signal
    }
    __syncthreads();
  }

  // --- epilogue: D layout lane n=l16, VGPR r -> m = r + 8*hs ---
  #pragma unroll
  for (int ai = 0; ai < 4; ++ai) {
    #pragma unroll
    for (int nj = 0; nj < 2; ++nj) {
      int col = nBlk + wN * 32 + nj * 16 + l16;
      float bv = bias[(size_t)e * N + col];
      #pragma unroll
      for (int r = 0; r < 8; ++r) {
        int mloc = wM * 64 + ai * 16 + r + 8 * hs;
        int gRow = mtile * 128 + mloc;
        if (gRow >= cnt) continue;           // ragged M-edge guard
        int p = base + gRow;
        float v = acc[ai][nj][r] + bv;
        if (isGemm1) {
          // exact GELU: 0.5*x*(1+erf(x/sqrt(2)))
          float g = 0.5f * v * (1.0f + erff(v * 0.70710678118654752f));
          dstBf[(size_t)p * N + col] = f2bf(g);
        } else {
          dstF[(size_t)p * N + col] = v * posw[p];
        }
      }
    }
  }
}

// ---------------------------------------------------------------------------
// 4) Combine: out = hidden + partial[slot0] + partial[slot1]
// ---------------------------------------------------------------------------
__global__ __launch_bounds__(256)
void combine_kernel(const float* __restrict__ hidden,
                    const float* __restrict__ partial, // [2T, H] (already weighted)
                    const int* __restrict__ tok2pos,   // [T, 2]
                    float* __restrict__ out)
{
  int i = blockIdx.x * blockDim.x + threadIdx.x;
  const int HV = HDIM / 4;
  if (i >= TDIM * HV) return;
  int t  = i / HV;
  int h4 = i - t * HV;
  int p0 = tok2pos[t * 2 + 0];
  int p1 = tok2pos[t * 2 + 1];
  float4 hv = ((const float4*)(hidden  + (size_t)t  * HDIM))[h4];
  float4 a  = ((const float4*)(partial + (size_t)p0 * HDIM))[h4];
  float4 b  = ((const float4*)(partial + (size_t)p1 * HDIM))[h4];
  float4 r;
  r.x = hv.x + a.x + b.x;
  r.y = hv.y + a.y + b.y;
  r.z = hv.z + a.z + b.z;
  r.w = hv.w + a.w + b.w;
  ((float4*)(out + (size_t)t * HDIM))[h4] = r;
}

// ---------------------------------------------------------------------------
// Launch
// ---------------------------------------------------------------------------
extern "C" void kernel_launch(void* const* d_in, const int* in_sizes, int n_in,
                              void* d_out, int out_size, void* d_ws, size_t ws_size,
                              hipStream_t stream) {
  const float* hidden = (const float*)d_in[0];   // [B,S,H]
  const float* gate_w = (const float*)d_in[1];   // [H,E]
  const float* w1     = (const float*)d_in[2];   // [E,H,F]
  const float* b1     = (const float*)d_in[3];   // [E,F]
  const float* w2     = (const float*)d_in[4];   // [E,F,H]
  const float* b2     = (const float*)d_in[5];   // [E,H]
  float* out = (float*)d_out;

  // Workspace carve-up (~437 MB total)
  size_t off = 0;
  auto carve = [&](size_t bytes) -> char* {
    char* p = (char*)d_ws + off;
    off += (bytes + 255) & ~(size_t)255;
    return p;
  };
  unsigned short* tokensBf = (unsigned short*)carve((size_t)TDIM * HDIM * 2);      // 32 MB
  unsigned short* hmid     = (unsigned short*)carve((size_t)2 * TDIM * FDIM * 2);  // 256 MB
  float* partial = (float*)carve((size_t)2 * TDIM * HDIM * 4);                     // 128 MB
  float* topw    = (float*)carve((size_t)TDIM * 2 * 4);
  int*   topidx  = (int*)  carve((size_t)TDIM * 2 * 4);
  int*   counts  = (int*)  carve(EXP * 4);
  int*   baseOff = (int*)  carve(EXP * 4);
  int*   cursor  = (int*)  carve(EXP * 4);
  int*   rowlist = (int*)  carve((size_t)2 * TDIM * 4);
  float* poswArr = (float*)carve((size_t)2 * TDIM * 4);
  int*   tok2pos = (int*)  carve((size_t)TDIM * 2 * 4);

  // 1) gate + bf16 token copy
  gate_kernel<<<TDIM, 256, 0, stream>>>(hidden, gate_w, topw, topidx, tokensBf);

  // 2) routing
  route_init<<<1, 32, 0, stream>>>(counts, cursor);
  route_count<<<(2 * TDIM + 255) / 256, 256, 0, stream>>>(topidx, counts);
  route_scan<<<1, 1, 0, stream>>>(counts, baseOff);
  route_scatter<<<(2 * TDIM + 255) / 256, 256, 0, stream>>>(
      topidx, topw, baseOff, cursor, rowlist, poswArr, tok2pos);

  // 3) GEMM1: hmid = gelu(tokens @ w1[e] + b1[e])   [M=cnt, K=H, N=F]
  moe_gemm_kernel<<<dim3(FDIM / 128, EXP * MTILES), 256, 0, stream>>>(
      tokensBf, w1, b1, counts, baseOff, rowlist, poswArr,
      hmid, partial, HDIM, FDIM, 1);

  // 4) GEMM2: partial = gatew * (hmid @ w2[e] + b2[e])   [M=cnt, K=F, N=H]
  moe_gemm_kernel<<<dim3(HDIM / 128, EXP * MTILES), 256, 0, stream>>>(
      hmid, w2, b2, counts, baseOff, rowlist, poswArr,
      hmid, partial, FDIM, HDIM, 0);

  // 5) combine + residual
  combine_kernel<<<(TDIM * (HDIM / 4) + 255) / 256, 256, 0, stream>>>(
      hidden, partial, tok2pos, out);
}
